// RoIOPLDHead_76579266887878
// MI455X (gfx1250) — matrix operand order repaired and checked by reading, not compiled, not run on hardware
//
#include <hip/hip_runtime.h>
#include <hip/hip_bf16.h>

// ---------------------------------------------------------------------------
// MI455X / gfx1250.  All GEMM-shaped work on v_wmma_f32_16x16x32_bf16.
// Activations are position-major [pos][ch] and ring-padded (18x18, 2-px zero
// halo), so each lane's WMMA B fragment is two 16-byte K runs: both GEMM
// kernels load fragments DIRECTLY from global memory (global_load_b128) --
// no LDS, no barriers.  Spatial taps (3x3 / 5x5) are branchless halo reads.
// ---------------------------------------------------------------------------

#define NGf   100352      // 512 * 196 interior positions
#define HWf   196
#define PP    324         // 18*18 padded positions per image
#define NPf   165888      // 512 * 324
#define NSTR  125440      // 640 * 196 (d_out image stride)

typedef __attribute__((ext_vector_type(16))) __bf16 v16bf;
typedef __attribute__((ext_vector_type(8)))  float  v8f;

struct U32x8 { uint4 x; uint4 y; };   // 32B carrier for a 16 x bf16 fragment

__device__ __forceinline__ unsigned short f2bf(float f) {
  unsigned u = __float_as_uint(f);
  u += 0x7FFFu + ((u >> 16) & 1u);            // round-to-nearest-even
  return (unsigned short)(u >> 16);
}
__device__ __forceinline__ float bf2f(unsigned short h) {
  return __uint_as_float(((unsigned)h) << 16);
}
__device__ __forceinline__ v8f wmma_bf16(v16bf a, v16bf b, v8f c) {
  return __builtin_amdgcn_wmma_f32_16x16x32_bf16(false, a, false, b,
                                                 (short)0, c, false, false);
}
__device__ __forceinline__ v16bf frag(uint4 lo, uint4 hi) {
  return __builtin_bit_cast(v16bf, U32x8{lo, hi});
}

// ---------------------------------------------------------------------------
// 3x3 conv as implicit GEMM, taps outer.
//   X  : [NP][IC]  ring-padded bf16, position-major (K contiguous!)
//   Wt : [9][OC][IC] bf16
//   Y  : [NG][OC]  f32, position-major
// Block 256 thr = 8 waves (2M x 4N); wave = 32x32 out via 2x2 WMMA tiles.
// No LDS, no barriers: all fragments via direct global_load_b128.
// ---------------------------------------------------------------------------
__global__ __launch_bounds__(256)
void k_conv3x3_wmma(const unsigned short* __restrict__ X,
                    const unsigned short* __restrict__ Wt,
                    const float* __restrict__ bias,
                    float* __restrict__ Y,
                    int IC, int OC)
{
  const int tid  = threadIdx.x;
  const int lane = tid & 31;
  const int wv   = tid >> 5;
  const int mBase = blockIdx.x * 64 + (wv & 1) * 32;
  const int nBase = blockIdx.y * 128 + (wv >> 1) * 32;

  // two 16-wide column groups handled by this wave
  const int g0 = nBase + (lane & 15);
  const int g1 = g0 + 16;
  int n0 = g0 / HWf, p0 = g0 - n0 * HWf, y0 = p0 / 14, x0 = p0 - y0 * 14;
  int n1 = g1 / HWf, p1 = g1 - n1 * HWf, y1 = p1 / 14, x1 = p1 - y1 * 14;
  const size_t pos0 = (size_t)n0 * PP + (y0 + 2) * 18 + (x0 + 2);
  const size_t pos1 = (size_t)n1 * PP + (y1 + 2) * 18 + (x1 + 2);

  const int abase = (lane < 16) ? 0 : 8;    // A half-lane K groups
  const int bbase = (lane < 16) ? 0 : 8;    // B half-lane K groups
  const int aRow  = mBase + (lane & 15);

  v8f acc00 = {}, acc01 = {}, acc10 = {}, acc11 = {};

  for (int t = 0; t < 9; ++t) {
    const long toff = (long)((t / 3 - 1) * 18 + (t % 3 - 1)) * IC; // halo-safe
    const unsigned short* a0p = Wt + ((size_t)t * OC + aRow) * IC + abase;
    const unsigned short* a1p = a0p + (size_t)16 * IC;
    const unsigned short* b0p = X + (long)pos0 * IC + toff + bbase;
    const unsigned short* b1p = X + (long)pos1 * IC + toff + bbase;

    for (int kb = 0; kb < IC; kb += 32) {
      uint4 a0l = *(const uint4*)(a0p + kb);
      uint4 a0h = *(const uint4*)(a0p + kb + 16);
      uint4 a1l = *(const uint4*)(a1p + kb);
      uint4 a1h = *(const uint4*)(a1p + kb + 16);
      uint4 b0l = *(const uint4*)(b0p + kb);
      uint4 b0h = *(const uint4*)(b0p + kb + 16);
      uint4 b1l = *(const uint4*)(b1p + kb);
      uint4 b1h = *(const uint4*)(b1p + kb + 16);
      if (kb + 32 < IC) __builtin_prefetch(b0p + kb + 32, 0, 0);

      v16bf aF0 = frag(a0l, a0h), aF1 = frag(a1l, a1h);
      v16bf bF0 = frag(b0l, b0h), bF1 = frag(b1l, b1h);
      acc00 = wmma_bf16(aF0, bF0, acc00);
      acc01 = wmma_bf16(aF0, bF1, acc01);
      acc10 = wmma_bf16(aF1, bF0, acc10);
      acc11 = wmma_bf16(aF1, bF1, acc11);
    }
  }

  const int mofs = (lane < 16) ? 0 : 8;
  float* yo0 = Y + (size_t)g0 * OC + mBase + mofs;
  float* yo1 = Y + (size_t)g1 * OC + mBase + mofs;
  #pragma unroll
  for (int r = 0; r < 8; ++r) {
    float bb0 = bias[mBase + mofs + r];
    float bb1 = bias[mBase + mofs + r + 16];
    yo0[r]      = acc00[r] + bb0;
    yo0[r + 16] = acc10[r] + bb1;
    yo1[r]      = acc01[r] + bb0;
    yo1[r + 16] = acc11[r] + bb1;
  }
}

// ---------------------------------------------------------------------------
// Pointwise 1x1 (64x64) GEMM.  B: [NG][64] position-major (K contiguous).
// mode 0: out[g*320 + chBase + oc] += v    (x_fo accumulator, pos-major)
// mode 1: out[n, chBase+oc, p]    += v     (x_so into d_out, NCHW)
// ---------------------------------------------------------------------------
__global__ __launch_bounds__(256)
void k_pw1x1_wmma(const unsigned short* __restrict__ A64,   // [64][64]
                  const unsigned short* __restrict__ B,     // [NG][64]
                  const float* __restrict__ bias,           // [64]
                  float* __restrict__ out,
                  int mode, int chBase)
{
  const int tid  = threadIdx.x;
  const int lane = tid & 31;
  const int wv   = tid >> 5;
  const int mBase = (wv & 1) * 32;
  const int nBase = blockIdx.y * 128 + (wv >> 1) * 32;

  const int g0 = nBase + (lane & 15);
  const int g1 = g0 + 16;
  const int abase = (lane < 16) ? 0 : 8;
  const int bbase = (lane < 16) ? 0 : 8;
  const int aRow  = mBase + (lane & 15);

  const unsigned short* a0p = A64 + (size_t)aRow * 64 + abase;
  const unsigned short* a1p = a0p + 16 * 64;
  const unsigned short* b0p = B + (size_t)g0 * 64 + bbase;
  const unsigned short* b1p = B + (size_t)g1 * 64 + bbase;

  v8f acc00 = {}, acc01 = {}, acc10 = {}, acc11 = {};

  #pragma unroll
  for (int kb = 0; kb < 64; kb += 32) {
    uint4 a0l = *(const uint4*)(a0p + kb);
    uint4 a0h = *(const uint4*)(a0p + kb + 16);
    uint4 a1l = *(const uint4*)(a1p + kb);
    uint4 a1h = *(const uint4*)(a1p + kb + 16);
    uint4 b0l = *(const uint4*)(b0p + kb);
    uint4 b0h = *(const uint4*)(b0p + kb + 16);
    uint4 b1l = *(const uint4*)(b1p + kb);
    uint4 b1h = *(const uint4*)(b1p + kb + 16);

    v16bf aF0 = frag(a0l, a0h), aF1 = frag(a1l, a1h);
    v16bf bF0 = frag(b0l, b0h), bF1 = frag(b1l, b1h);
    acc00 = wmma_bf16(aF0, bF0, acc00);
    acc01 = wmma_bf16(aF0, bF1, acc01);
    acc10 = wmma_bf16(aF1, bF0, acc10);
    acc11 = wmma_bf16(aF1, bF1, acc11);
  }

  const int mofs = (lane < 16) ? 0 : 8;
  if (mode == 0) {
    float* o0 = out + (size_t)g0 * 320 + chBase + mBase + mofs;
    float* o1 = out + (size_t)g1 * 320 + chBase + mBase + mofs;
    #pragma unroll
    for (int r = 0; r < 8; ++r) {
      float b0 = bias[mBase + mofs + r];
      float b1 = bias[mBase + mofs + r + 16];
      o0[r]      += acc00[r] + b0;
      o0[r + 16] += acc10[r] + b1;
      o1[r]      += acc01[r] + b0;
      o1[r + 16] += acc11[r] + b1;
    }
  } else {
    int n0 = g0 / HWf, p0 = g0 - n0 * HWf;
    int n1 = g1 / HWf, p1 = g1 - n1 * HWf;
    #pragma unroll
    for (int r = 0; r < 8; ++r) {
      int oc0 = mBase + mofs + r;
      int oc1 = oc0 + 16;
      float b0 = bias[oc0], b1 = bias[oc1];
      out[(size_t)n0 * NSTR + (size_t)(chBase + oc0) * HWf + p0] += acc00[r] + b0;
      out[(size_t)n1 * NSTR + (size_t)(chBase + oc0) * HWf + p1] += acc01[r] + b0;
      out[(size_t)n0 * NSTR + (size_t)(chBase + oc1) * HWf + p0] += acc10[r] + b1;
      out[(size_t)n1 * NSTR + (size_t)(chBase + oc1) * HWf + p1] += acc11[r] + b1;
    }
  }
}

// ---------------------------------------------------------------------------
// GroupNorm over pos-major [NG][320]; group = 10 consecutive channels.
// ---------------------------------------------------------------------------
__global__ __launch_bounds__(256)
void k_gn_stats(const float* __restrict__ A, float* __restrict__ stats)
{
  const int n = blockIdx.x, grp = blockIdx.y;
  __shared__ float s1[256], s2[256];
  float a = 0.f, b = 0.f;
  for (int i = threadIdx.x; i < 10 * HWf; i += 256) {
    int p = i / 10, c = i - p * 10;     // channel-fastest
    float v = A[((size_t)n * HWf + p) * 320 + grp * 10 + c];
    a += v; b += v * v;
  }
  s1[threadIdx.x] = a; s2[threadIdx.x] = b;
  __syncthreads();
  for (int s = 128; s > 0; s >>= 1) {
    if (threadIdx.x < s) { s1[threadIdx.x] += s1[threadIdx.x + s];
                           s2[threadIdx.x] += s2[threadIdx.x + s]; }
    __syncthreads();
  }
  if (threadIdx.x == 0) {
    float m   = s1[0] * (1.0f / 1960.0f);
    float var = s2[0] * (1.0f / 1960.0f) - m * m;
    stats[(n * 32 + grp) * 2 + 0] = m;
    stats[(n * 32 + grp) * 2 + 1] = rsqrtf(var + 1e-5f);
  }
}

// normalize + ReLU: A [NG][320] -> hbP [NP][320] padded bf16 (+ f32 NCHW h)
__global__ __launch_bounds__(256)
void k_gn_apply(const float* __restrict__ A, const float* __restrict__ stats,
                const float* __restrict__ gamma, const float* __restrict__ beta,
                unsigned short* __restrict__ hbP, float* __restrict__ dout)
{
  long i = (long)blockIdx.x * 256 + threadIdx.x;      // NG*320 exact
  int g  = (int)(i / 320);
  int ch = (int)(i - (long)g * 320);
  int n  = g / HWf, p = g - n * HWf;
  int y  = p / 14,  x = p - y * 14;
  int grp = ch / 10;
  float m  = stats[(n * 32 + grp) * 2 + 0];
  float rs = stats[(n * 32 + grp) * 2 + 1];
  float v  = (A[i] - m) * rs * gamma[ch] + beta[ch];
  v = fmaxf(v, 0.f);
  hbP[((size_t)n * PP + (y + 2) * 18 + (x + 2)) * 320 + ch] = f2bf(v);
  if (dout) dout[(size_t)n * NSTR + (size_t)ch * HWf + p] = v;
}

// ---------------------------------------------------------------------------
// Depthwise 5x5 on ring-padded pos-major input (halo zeros -> branchless).
// Block = 4 positions x 64 channels (channel-fastest, coalesced).
// ---------------------------------------------------------------------------
__global__ __launch_bounds__(256)
void k_dw5x5(const unsigned short* __restrict__ XinP,  // [NP][RC]
             int RC, int chBase,
             const float* __restrict__ W25, const float* __restrict__ db,
             unsigned short* __restrict__ outB)        // [NG][64]
{
  const int tid = threadIdx.x;
  const int c   = tid & 63;
  const int g   = blockIdx.x * 4 + (tid >> 6);     // grid.x = NG/4
  const int n = g / HWf, p = g - n * HWf;
  const int y = p / 14,  x = p - y * 14;
  const unsigned short* src =
      XinP + ((size_t)n * PP + y * 18 + x) * RC + chBase + c; // window top-left
  const float* wk = W25 + c * 25;
  float acc = db[c];
  #pragma unroll
  for (int dy = 0; dy < 5; ++dy)
    #pragma unroll
    for (int dx = 0; dx < 5; ++dx)
      acc += bf2f(src[((size_t)dy * 18 + dx) * RC]) * wk[dy * 5 + dx];
  outB[(size_t)g * 64 + c] = f2bf(acc);
}

// ---------------------------------------------------------------------------
// Conversion / layout helpers.
// ---------------------------------------------------------------------------
__global__ void k_zero16(unsigned short* __restrict__ d, long n)
{ long i = (long)blockIdx.x * 256 + threadIdx.x; if (i < n) d[i] = 0; }

// NCHW f32 x -> ring-padded pos-major bf16 [NP][256] (halo = 0)
__global__ void k_cvt_x(const float* __restrict__ x, unsigned short* __restrict__ XbP)
{
  long i = (long)blockIdx.x * 256 + threadIdx.x;     // NP*256 exact
  int gp = (int)(i >> 8);
  int ic = (int)(i & 255);
  int n  = gp / PP, r = gp - n * PP;
  int yy = r / 18, xx = r - yy * 18;
  unsigned short v = 0;
  if (yy >= 2 && yy < 16 && xx >= 2 && xx < 16)
    v = f2bf(x[((size_t)n * 256 + ic) * HWf + (yy - 2) * 14 + (xx - 2)]);
  XbP[i] = v;
}

// OIHW f32 -> [tap][oc][ic] bf16
__global__ void k_cvt_w(const float* __restrict__ src, unsigned short* __restrict__ dst,
                        int OC, int IC)
{
  int i = blockIdx.x * 256 + threadIdx.x;
  int total = 9 * OC * IC;
  if (i >= total) return;
  int t  = i / (OC * IC);
  int r  = i - t * OC * IC;
  int oc = r / IC, ic = r - oc * IC;
  dst[i] = f2bf(src[((size_t)oc * IC + ic) * 9 + t]);
}

__global__ void k_f2b(const float* __restrict__ s, unsigned short* __restrict__ d, long n)
{ long i = (long)blockIdx.x * 256 + threadIdx.x; if (i < n) d[i] = f2bf(s[i]); }

// x_fo init: padded bf16 h -> pos-major f32 [NG][320]
__global__ void k_h2fo(const unsigned short* __restrict__ hbP, float* __restrict__ ffo)
{
  long i = (long)blockIdx.x * 256 + threadIdx.x;     // NG*320 exact
  int g  = (int)(i / 320);
  int ch = (int)(i - (long)g * 320);
  int n  = g / HWf, p = g - n * HWf;
  int y  = p / 14,  x = p - y * 14;
  ffo[i] = bf2f(hbP[((size_t)n * PP + (y + 2) * 18 + (x + 2)) * 320 + ch]);
}

// x_fo f32 [NG][320] -> ring-padded pos-major bf16 [NP][320] (full overwrite)
__global__ void k_cvt_fob(const float* __restrict__ ffo, unsigned short* __restrict__ dst)
{
  long i = (long)blockIdx.x * 256 + threadIdx.x;     // NP*320 exact
  int gp = (int)(i / 320);
  int ch = (int)(i - (long)gp * 320);
  int n  = gp / PP, r = gp - n * PP;
  int yy = r / 18, xx = r - yy * 18;
  unsigned short v = 0;
  if (yy >= 2 && yy < 16 && xx >= 2 && xx < 16)
    v = f2bf(ffo[((size_t)n * HWf + (yy - 2) * 14 + (xx - 2)) * 320 + ch]);
  dst[i] = v;
}

// x_so base = h : out[:,320:640] = out[:,0:320]
__global__ void k_xso_init(float* __restrict__ out)
{
  long i = (long)blockIdx.x * 256 + threadIdx.x;     // 512*62720 exact
  long n = i / 62720;
  long r = i - n * 62720;
  out[n * NSTR + 62720 + r] = out[n * NSTR + r];
}

// ---------------------------------------------------------------------------
extern "C" void kernel_launch(void* const* d_in, const int* in_sizes, int n_in,
                              void* d_out, int out_size, void* d_ws, size_t ws_size,
                              hipStream_t stream)
{
  (void)in_sizes; (void)n_in; (void)out_size; (void)ws_size;

  const float* x       = (const float*)d_in[0];
  const float* conv0_w = (const float*)d_in[1];
  const float* conv0_b = (const float*)d_in[2];
  const float* gn0_g   = (const float*)d_in[3];
  const float* gn0_b   = (const float*)d_in[4];
  const float* convs_w = (const float*)d_in[5];
  const float* convs_b = (const float*)d_in[6];
  const float* gns_g   = (const float*)d_in[7];
  const float* gns_b   = (const float*)d_in[8];
  const float* fo_dw   = (const float*)d_in[9];
  const float* fo_db   = (const float*)d_in[10];
  const float* fo_pw   = (const float*)d_in[11];
  const float* fo_pb   = (const float*)d_in[12];
  const float* so_dw   = (const float*)d_in[13];
  const float* so_db   = (const float*)d_in[14];
  const float* so_pw   = (const float*)d_in[15];
  const float* so_pb   = (const float*)d_in[16];
  float* out = (float*)d_out;

  const size_t NG = 100352, NP = 165888;
  char* wsb = (char*)d_ws;
  size_t o = 0;
  auto take = [&](size_t bytes) -> char* {
    char* p = wsb + o;
    o += (bytes + 255) & ~(size_t)255;
    return p;
  };
  // PpadA: x bf16 padded [NP][256] during tower, then x_fo bf16 padded [NP][320]
  unsigned short* PpadA = (unsigned short*)take(NP * 320 * 2);
  unsigned short* hbP   = (unsigned short*)take(NP * 320 * 2);   // h bf16 padded
  unsigned short* Wc    = (unsigned short*)take((size_t)(9*320*256 + 3*9*320*320)*2);
  unsigned short* PWb   = (unsigned short*)take((size_t)32 * 4096 * 2);
  float*          actF  = (float*)take(NG * 320 * 4);  // conv f32 out / x_fo acc
  float*          st    = (float*)take((size_t)512 * 32 * 2 * 4);
  unsigned short* dwt   = (unsigned short*)take(NG * 64 * 2);

  // ---- init / conversions -------------------------------------------------
  k_zero16<<<(int)(NP * 320 / 256), 256, 0, stream>>>(hbP, (long)NP * 320);
  k_cvt_x<<<(int)NP, 256, 0, stream>>>(x, PpadA);
  k_cvt_w<<<(9*320*256 + 255) / 256, 256, 0, stream>>>(conv0_w, Wc, 320, 256);
  for (int k = 0; k < 3; ++k)
    k_cvt_w<<<(9*320*320 + 255) / 256, 256, 0, stream>>>(
        convs_w + (size_t)k * 320 * 320 * 9,
        Wc + 737280 + (size_t)k * 921600, 320, 320);
  k_f2b<<<(16*64*64 + 255) / 256, 256, 0, stream>>>(fo_pw, PWb,           16*64*64);
  k_f2b<<<(16*64*64 + 255) / 256, 256, 0, stream>>>(so_pw, PWb + 16*4096, 16*64*64);

  // ---- conv tower: 4 x (conv3x3 WMMA -> GN -> ReLU) -----------------------
  const size_t wofs[4] = {0, 737280, 737280 + 921600, 737280 + 2*921600};
  for (int layer = 0; layer < 4; ++layer) {
    const unsigned short* in = (layer == 0) ? PpadA : hbP;
    const int IC = (layer == 0) ? 256 : 320;
    const float* bias = (layer == 0) ? conv0_b : convs_b + (layer - 1) * 320;
    const float* gg   = (layer == 0) ? gn0_g   : gns_g   + (layer - 1) * 320;
    const float* gb   = (layer == 0) ? gn0_b   : gns_b   + (layer - 1) * 320;

    k_conv3x3_wmma<<<dim3(5, 784), 256, 0, stream>>>(in, Wc + wofs[layer],
                                                     bias, actF, IC, 320);
    k_gn_stats<<<dim3(512, 32), 256, 0, stream>>>(actF, st);
    k_gn_apply<<<(int)(NG * 320 / 256), 256, 0, stream>>>(
        actF, st, gg, gb, hbP, (layer == 3) ? out : nullptr);
  }

  // ---- message passing ----------------------------------------------------
  static const int tgt [16] = {0,0,0, 1,1,1, 2,2,2, 3,3,3, 4,4,4,4};
  static const int srcp[16] = {1,2,4, 0,3,4, 0,3,4, 1,2,4, 0,1,2,3};

  k_xso_init<<<(int)(512 * 62720 / 256), 256, 0, stream>>>(out);
  k_h2fo<<<(int)(NG * 320 / 256), 256, 0, stream>>>(hbP, actF);

  // first order: x_fo[i] += pw(dw(h[src]))
  for (int e = 0; e < 16; ++e) {
    k_dw5x5<<<(int)(NG / 4), 256, 0, stream>>>(hbP, 320, srcp[e] * 64,
                                               fo_dw + (size_t)e * 64 * 25,
                                               fo_db + e * 64, dwt);
    k_pw1x1_wmma<<<dim3(1, 784), 256, 0, stream>>>(
        PWb + (size_t)e * 4096, dwt, fo_pb + e * 64,
        actF, 0, tgt[e] * 64);
  }

  // x_fo -> padded pos-major bf16 (reuses PpadA; full overwrite incl. halo)
  k_cvt_fob<<<(int)(NP * 320 / 256), 256, 0, stream>>>(actF, PpadA);

  // second order: out[:, 320+i*64 ..] += pw(dw(x_fo[src]))
  for (int e = 0; e < 16; ++e) {
    k_dw5x5<<<(int)(NG / 4), 256, 0, stream>>>(PpadA, 320, srcp[e] * 64,
                                               so_dw + (size_t)e * 64 * 25,
                                               so_db + e * 64, dwt);
    k_pw1x1_wmma<<<dim3(1, 784), 256, 0, stream>>>(
        PWb + (size_t)(16 + e) * 4096, dwt, so_pb + e * 64,
        out, 1, 320 + tgt[e] * 64);
  }
}
